// Wav2Vec2BertEncoderLayer_23424751632755
// MI455X (gfx1250) — matrix operand order repaired and checked by reading, not compiled, not run on hardware
//
#include <hip/hip_runtime.h>
#include <cstdint>
#include <cstddef>

typedef unsigned short u16;
typedef unsigned int   u32;
typedef __attribute__((ext_vector_type(16))) __bf16 v16bf;
typedef __attribute__((ext_vector_type(8)))  float  v8f;
typedef __attribute__((ext_vector_type(8)))  unsigned int v8u;

// ---------------- bf16 helpers (bit-level, no __bf16 arithmetic) -------------
__device__ __forceinline__ u16 f2bf(float f) {
    u32 u = __float_as_uint(f);
    u32 r = u + 0x7FFFu + ((u >> 16) & 1u);   // round-to-nearest-even
    return (u16)(r >> 16);
}
__device__ __forceinline__ float bf2f(u16 h) {
    return __uint_as_float(((u32)h) << 16);
}
__device__ __forceinline__ float silu_f(float x) { return x / (1.0f + __expf(-x)); }

// ---------------- CDNA5 async global->LDS copy (ASYNCcnt-tracked) ------------
// LDS dest = per-lane 32-bit LDS byte offset (low 32 bits of a flat shared ptr,
// per the aperture layout ADDR[31:0] = LDS offset). 16 bytes per lane.
__device__ __forceinline__ u32 lds_off32(const void* p) {
    return (u32)(size_t)p;
}
__device__ __forceinline__ void async_copy_b128(u32 lds_dst, const void* gsrc) {
    asm volatile("global_load_async_to_lds_b128 %0, %1, off"
                 :: "v"(lds_dst), "v"(gsrc)
                 : "memory");
}
__device__ __forceinline__ void wait_async0() {
    asm volatile("s_wait_asynccnt 0" ::: "memory");
}

// ---------------- WMMA wrapper ----------------------------------------------
__device__ __forceinline__ v8f wmma_bf16(const v8u& a, const v8u& b, const v8f& c) {
    return __builtin_amdgcn_wmma_f32_16x16x32_bf16(
        false, __builtin_bit_cast(v16bf, a),
        false, __builtin_bit_cast(v16bf, b),
        (short)0, c, false, false);
}

// A fragment 16x32 bf16: lane l -> row m = m_base + (l&15), g = l>>4
// dword j holds K pair: k = (2j<8 ? g*8+2j : 16+g*8+(2j-8))  [ISA 16-bit A layout]
__device__ __forceinline__ v8u load_afrag(const u16* base, int pitch, int m_base, int k_base) {
    const int l = threadIdx.x & 31;
    const int g = l >> 4;
    const u16* row = base + (size_t)(m_base + (l & 15)) * pitch + k_base;
    v8u r;
#pragma unroll
    for (int j = 0; j < 8; ++j) {
        int e = 2 * j;
        int k = (e < 8) ? (g * 8 + e) : (16 + g * 8 + (e - 8));
        r[j] = *(const u32*)(row + k);
    }
    return r;
}
// B fragment 32x16 bf16 stored n-major [n][k]: lane l -> col n = n_base + (l&15)
// dword j holds K pair: k = (2j<8 ? g*8+2j : 16+g*8+(2j-8))  (same as A layout)
__device__ __forceinline__ v8u load_bfrag(const u16* base, int pitch, int n_base, int k_base) {
    const int l = threadIdx.x & 31;
    const int g = l >> 4;
    const u16* row = base + (size_t)(n_base + (l & 15)) * pitch + k_base;
    v8u r;
#pragma unroll
    for (int j = 0; j < 8; ++j) {
        int e = 2 * j;
        int k = (e < 8) ? (g * 8 + e) : (16 + g * 8 + (e - 8));
        r[j] = *(const u32*)(row + k);
    }
    return r;
}

// ================= weight conversion kernels =================================
__global__ __launch_bounds__(256) void convert_transpose_kernel(const float* __restrict__ src,
                                                                u16* __restrict__ dst,
                                                                int K, int N) {
    // src: [K][N] f32 row-major  ->  dst: [N][K] bf16
    size_t idx = (size_t)blockIdx.x * 256 + threadIdx.x;
    if (idx >= (size_t)N * K) return;
    int n = (int)(idx / K);
    int k = (int)(idx % K);
    dst[idx] = f2bf(src[(size_t)k * N + n]);
}
__global__ __launch_bounds__(256) void convert_kernel(const float* __restrict__ src,
                                                      u16* __restrict__ dst, size_t count) {
    size_t idx = (size_t)blockIdx.x * 256 + threadIdx.x;
    if (idx < count) dst[idx] = f2bf(src[idx]);
}

// ================= row LayerNorm (D = 1024) ==================================
__global__ __launch_bounds__(256) void ln_kernel(const float* __restrict__ x,
                                                 const float* __restrict__ g,
                                                 const float* __restrict__ b,
                                                 void* __restrict__ out,
                                                 int act, int out_bf16) {
    __shared__ float r1[256];
    __shared__ float r2[256];
    const int t = threadIdx.x;
    const size_t row = blockIdx.x;
    const float* xr = x + row * 1024;
    float v[4];
    float s = 0.f, s2 = 0.f;
#pragma unroll
    for (int i = 0; i < 4; ++i) {
        v[i] = xr[t + i * 256];
        s += v[i];
        s2 += v[i] * v[i];
    }
    r1[t] = s; r2[t] = s2;
    __syncthreads();
    for (int off = 128; off > 0; off >>= 1) {
        if (t < off) { r1[t] += r1[t + off]; r2[t] += r2[t + off]; }
        __syncthreads();
    }
    float mu  = r1[0] * (1.f / 1024.f);
    float var = r2[0] * (1.f / 1024.f) - mu * mu;
    float inv = rsqrtf(var + 1e-5f);
#pragma unroll
    for (int i = 0; i < 4; ++i) {
        int c = t + i * 256;
        float y = (v[i] - mu) * inv * g[c] + b[c];
        if (act) y = silu_f(y);
        if (out_bf16) ((u16*)out)[row * 1024 + c] = f2bf(y);
        else          ((float*)out)[row * 1024 + c] = y;
    }
}

// ================= main GEMM: C = A(bf16 MxK) * B(bf16 [N][K])^T =============
struct GemmArgs {
    const u16*  A;      // [M][K] bf16
    const u16*  Bw;     // [N][K] bf16 (N-major)
    const float* bias;  // [N] or null
    const float* resid; // [M][N] f32 or null (mode 0)
    float* outf;        // mode 0
    u16*   outh;        // mode 1
    u16*   oq; u16* ok; u16* ov; // mode 2 (qkv scatter)
    int M, N, K;
    int act;    // 1 = silu
    int mode;   // 0: f32 out = val*alpha + resid ; 1: bf16 out ; 2: qkv scatter
    float alpha;
};

#define GP 40  // LDS pitch (elements): 80B rows -> 16B-aligned + conflict-free frag reads

__global__ __launch_bounds__(256) void gemm_bf16_kernel(GemmArgs ga) {
    // double-buffered tiles, filled by async global->LDS DMA
    __shared__ __align__(16) u16 sA[2][128 * GP];
    __shared__ __align__(16) u16 sB[2][128 * GP];
    const int t    = threadIdx.x;
    const int n0   = blockIdx.x * 128;
    const int m0   = blockIdx.y * 128;
    const int wave = t >> 5;
    const int wm   = (wave >> 2) * 64;  // 2 waves along M
    const int wn   = (wave & 3) * 32;   // 4 waves along N

    v8f acc[4][2];
#pragma unroll
    for (int a = 0; a < 4; ++a)
#pragma unroll
        for (int b = 0; b < 2; ++b)
#pragma unroll
            for (int j = 0; j < 8; ++j) acc[a][b][j] = 0.0f;

    // stage one 128x32 A tile + 128x32 B tile into buffer `buf` (async, 16B/lane x2)
    auto stage = [&](int buf, int k0) {
#pragma unroll
        for (int i = 0; i < 2; ++i) {
            int c   = t + i * 256;          // 512 chunks of 16B per tile
            int r   = c >> 2;               // tile row
            int off = (c & 3) * 8;          // element offset in row
            async_copy_b128(lds_off32(&sA[buf][r * GP + off]),
                            ga.A + (size_t)(m0 + r) * ga.K + k0 + off);
            int n = n0 + r;
            if (n < ga.N)
                async_copy_b128(lds_off32(&sB[buf][r * GP + off]),
                                ga.Bw + (size_t)n * ga.K + k0 + off);
        }
    };

    const int nk = ga.K >> 5;
    stage(0, 0);
    wait_async0();
    __syncthreads();

    for (int ki = 0; ki < nk; ++ki) {
        const int buf = ki & 1;
        if (ki + 1 < nk) stage(buf ^ 1, (ki + 1) << 5);   // overlap DMA with WMMA

        v8u bf0 = load_bfrag(sB[buf], GP, wn, 0);
        v8u bf1 = load_bfrag(sB[buf], GP, wn + 16, 0);
#pragma unroll
        for (int mt = 0; mt < 4; ++mt) {
            v8u af = load_afrag(sA[buf], GP, wm + mt * 16, 0);
            acc[mt][0] = wmma_bf16(af, bf0, acc[mt][0]);
            acc[mt][1] = wmma_bf16(af, bf1, acc[mt][1]);
        }
        wait_async0();      // next tile landed in LDS
        __syncthreads();    // all waves done reading current tile
    }

    // epilogue: C layout lane l -> n = (l&15), m = (l>>4)*8 + j
    const int l = t & 31;
    const int half = l >> 4;
#pragma unroll
    for (int mt = 0; mt < 4; ++mt) {
#pragma unroll
        for (int nt = 0; nt < 2; ++nt) {
            int n = n0 + wn + nt * 16 + (l & 15);
            if (n >= ga.N) continue;
            float bb = ga.bias ? ga.bias[n] : 0.0f;
#pragma unroll
            for (int j = 0; j < 8; ++j) {
                int m = m0 + wm + mt * 16 + half * 8 + j;
                float v = acc[mt][nt][j] + bb;
                if (ga.act) v = silu_f(v);
                if (ga.mode == 0) {
                    float r = ga.resid ? ga.resid[(size_t)m * ga.N + n] : 0.0f;
                    ga.outf[(size_t)m * ga.N + n] = v * ga.alpha + r;
                } else if (ga.mode == 1) {
                    ga.outh[(size_t)m * ga.N + n] = f2bf(v);
                } else { // qkv scatter -> [B,H,S,64] bf16
                    int sec = n >> 10;
                    int h   = (n & 1023) >> 6;
                    int hd  = n & 63;
                    int b   = m >> 10;
                    int s   = m & 1023;
                    u16* dst = (sec == 0) ? ga.oq : (sec == 1 ? ga.ok : ga.ov);
                    dst[(((size_t)b * 16 + h) * 1024 + s) * 64 + hd] = f2bf(v);
                }
            }
        }
    }
}

// ================= flash attention with relative-position bias ===============
// grid: (S/64, B*H), block: 128 threads (4 waves), each wave owns 16 query rows
#define FP 72   // LDS pitch (elements): 144B rows, 16B-aligned, conflict-free

__global__ __launch_bounds__(128) void flash_attn_kernel(const u16* __restrict__ qb,
                                                         const u16* __restrict__ kb,
                                                         const u16* __restrict__ vb,
                                                         const u16* __restrict__ pe,
                                                         u16* __restrict__ ob) {
    __shared__ __align__(16) u16 sMem[144 * FP];     // phase1: PE[144][FP]; phase2: K[64][FP]+V[64][FP]
    __shared__ __align__(16) u16 sPB[4 * 16 * 132];  // per-wave q.pe bias (bf16), 129 cols pad 132
    __shared__ __align__(16) u16 sP[4 * 16 * FP];    // per-wave P tile (16 rows x 64 keys)

    const int t    = threadIdx.x;
    const int w    = t >> 5;
    const int l    = t & 31;
    const int half = l >> 4;
    const int bh   = blockIdx.y;        // b*16 + h
    const int q0   = blockIdx.x * 64;
    const int qrl  = w * 16 + (l & 15); // query row within block

    // --- phase 1: PE table -> LDS via async DMA; zero the pad rows ---
    u16* sPE = sMem;
    for (int i = t; i < 1032; i += 128) {            // 129 rows x 8 chunks(16B)
        int p   = i >> 3;
        int off = (i & 7) * 8;
        async_copy_b128(lds_off32(&sPE[p * FP + off]), pe + (size_t)p * 64 + off);
    }
    for (int i = t; i < 540; i += 128) {             // rows 129..143 zero (15 x 36 dwords)
        int p = 129 + i / 36;
        int c = i % 36;
        *(u32*)&sPE[p * FP + c * 2] = 0u;
    }

    // --- Q fragments (pre-scaled by HD^-0.5 = 0.125; exact in bf16) ---
    const u16* qrow = qb + ((size_t)bh * 1024 + q0 + qrl) * 64;
    v8u qf[2];
#pragma unroll
    for (int c = 0; c < 2; ++c) {
#pragma unroll
        for (int j = 0; j < 8; ++j) {
            int e = 2 * j;
            int k = ((e < 8) ? (half * 8 + e) : (16 + half * 8 + (e - 8))) + 32 * c;
            u32 d = *(const u32*)(qrow + k);
            float lo = bf2f((u16)(d & 0xffffu)) * 0.125f;
            float hi = bf2f((u16)(d >> 16)) * 0.125f;
            qf[c][j] = (u32)f2bf(lo) | ((u32)f2bf(hi) << 16);
        }
    }
    wait_async0();
    __syncthreads();

    // --- phase 1b: pbias[m][p] = q_m . pe_p  (WMMA, 9 n-tiles of 16 = 144 cols) ---
    u16* sPBw = sPB + w * 16 * 132;
#pragma unroll
    for (int nt = 0; nt < 9; ++nt) {
        v8f pc;
#pragma unroll
        for (int j = 0; j < 8; ++j) pc[j] = 0.f;
        pc = wmma_bf16(qf[0], load_bfrag(sPE, FP, nt * 16, 0), pc);
        pc = wmma_bf16(qf[1], load_bfrag(sPE, FP, nt * 16, 32), pc);
        int col = nt * 16 + (l & 15);
#pragma unroll
        for (int j = 0; j < 8; ++j) {
            if (col < 132) sPBw[(half * 8 + j) * 132 + col] = f2bf(pc[j]);
        }
    }
    __syncthreads();   // done with sPE; region reused for K/V below

    u16* sK  = sMem;
    u16* sV  = sMem + 64 * FP;
    u16* sPw = sP + w * 16 * FP;

    v8f O[4];
#pragma unroll
    for (int nt = 0; nt < 4; ++nt)
#pragma unroll
        for (int j = 0; j < 8; ++j) O[nt][j] = 0.f;
    float rM[8], rL[8];
#pragma unroll
    for (int j = 0; j < 8; ++j) { rM[j] = -3e38f; rL[j] = 0.f; }

    for (int kblk = 0; kblk < 16; ++kblk) {
        const int r0 = kblk * 64;
        // K tile [r][hd]: straight rows -> async DMA (512 x 16B chunks)
        for (int i = t; i < 512; i += 128) {
            int r   = i >> 3;
            int off = (i & 7) * 8;
            async_copy_b128(lds_off32(&sK[r * FP + off]),
                            kb + ((size_t)bh * 1024 + r0 + r) * 64 + off);
        }
        // V tile transposed [hd][r]: element scatter via DS
        for (int i = t; i < 2048; i += 128) {
            int r  = i >> 5;
            int kk = (i & 31) << 1;
            u32 dv = *(const u32*)(vb + ((size_t)bh * 1024 + r0 + r) * 64 + kk);
            sV[kk * FP + r]       = (u16)(dv & 0xffffu);
            sV[(kk + 1) * FP + r] = (u16)(dv >> 16);
        }
        wait_async0();
        __syncthreads();

        // scores S = (q/8) . k + pbias[clip(r-l)]
        v8f sf[4];
#pragma unroll
        for (int nt = 0; nt < 4; ++nt) {
            v8f s;
#pragma unroll
            for (int j = 0; j < 8; ++j) s[j] = 0.f;
            s = wmma_bf16(qf[0], load_bfrag(sK, FP, nt * 16, 0), s);
            s = wmma_bf16(qf[1], load_bfrag(sK, FP, nt * 16, 32), s);
            int rg = r0 + nt * 16 + (l & 15);
#pragma unroll
            for (int j = 0; j < 8; ++j) {
                int m  = half * 8 + j;
                int lq = q0 + w * 16 + m;
                int dd = rg - lq;
                dd = (dd < -64) ? -64 : (dd > 64 ? 64 : dd);
                s[j] += bf2f(sPBw[m * 132 + dd + 64]);
            }
            sf[nt] = s;
        }

        // online softmax per query row (16 lanes per half hold a row's columns)
#pragma unroll
        for (int j = 0; j < 8; ++j) {
            float mx = fmaxf(fmaxf(sf[0][j], sf[1][j]), fmaxf(sf[2][j], sf[3][j]));
            mx = fmaxf(mx, __shfl_xor(mx, 1));
            mx = fmaxf(mx, __shfl_xor(mx, 2));
            mx = fmaxf(mx, __shfl_xor(mx, 4));
            mx = fmaxf(mx, __shfl_xor(mx, 8));
            float nm = fmaxf(rM[j], mx);
            float fs = __expf(rM[j] - nm);
            rM[j] = nm;
            rL[j] *= fs;
#pragma unroll
            for (int nt = 0; nt < 4; ++nt) O[nt][j] *= fs;
            float ls = 0.f;
            int m = half * 8 + j;
#pragma unroll
            for (int nt = 0; nt < 4; ++nt) {
                float p = __expf(sf[nt][j] - nm);
                ls += p;
                sPw[m * FP + nt * 16 + (l & 15)] = f2bf(p);  // P -> LDS (A layout source)
            }
            ls += __shfl_xor(ls, 1);
            ls += __shfl_xor(ls, 2);
            ls += __shfl_xor(ls, 4);
            ls += __shfl_xor(ls, 8);
            rL[j] += ls;
        }

        // O += P x V   (per-wave LDS; ds ops complete in-order within a wave)
        v8u pa0 = load_afrag(sPw, FP, 0, 0);
        v8u pa1 = load_afrag(sPw, FP, 0, 32);
#pragma unroll
        for (int nt = 0; nt < 4; ++nt) {
            O[nt] = wmma_bf16(pa0, load_bfrag(sV, FP, nt * 16, 0), O[nt]);
            O[nt] = wmma_bf16(pa1, load_bfrag(sV, FP, nt * 16, 32), O[nt]);
        }
        __syncthreads();  // guard K/V reuse next iteration
    }

    // epilogue: write O/rowsum as bf16 into [B,S,D] (col = h*64 + hd)
    const int b = bh >> 4, h = bh & 15;
#pragma unroll
    for (int nt = 0; nt < 4; ++nt) {
#pragma unroll
        for (int j = 0; j < 8; ++j) {
            int m    = half * 8 + j;
            int srow = q0 + w * 16 + m;
            float v  = O[nt][j] / rL[j];
            ob[((size_t)b * 1024 + srow) * 1024 + h * 64 + nt * 16 + (l & 15)] = f2bf(v);
        }
    }
}

// ================= GLU: a * sigmoid(g) over channel split ===================
__global__ __launch_bounds__(256) void glu_kernel(const float* __restrict__ gin,
                                                  u16* __restrict__ out) {
    size_t idx = (size_t)blockIdx.x * 256 + threadIdx.x;
    size_t row = idx >> 10;
    int c = (int)(idx & 1023);
    float a = gin[row * 2048 + c];
    float g = gin[row * 2048 + 1024 + c];
    out[idx] = f2bf(a * (1.f / (1.f + __expf(-g))));
}

// ================= causal depthwise conv (K = 31) ===========================
// grid: (D/256, S/128, B); each thread: one channel, 128 timesteps, register window
__global__ __launch_bounds__(256) void dwconv_kernel(const u16* __restrict__ cin,
                                                     const float* __restrict__ w,
                                                     float* __restrict__ y) {
    const int d  = blockIdx.x * 256 + threadIdx.x;
    const int s0 = blockIdx.y * 128;
    const int b  = blockIdx.z;
    float wv[31];
#pragma unroll
    for (int t = 0; t < 31; ++t) wv[t] = w[d * 31 + t];
    float win[31];
#pragma unroll
    for (int t = 0; t < 31; ++t) {
        int s = s0 - 30 + t;
        win[t] = (s >= 0) ? bf2f(cin[((size_t)b * 1024 + s) * 1024 + d]) : 0.f;
    }
    for (int i = 0; i < 128; ++i) {
        int s = s0 + i;
        float acc = 0.f;
#pragma unroll
        for (int t = 0; t < 31; ++t) acc += wv[t] * win[t];
        y[((size_t)b * 1024 + s) * 1024 + d] = acc;
#pragma unroll
        for (int t = 0; t < 30; ++t) win[t] = win[t + 1];
        int sn = s + 1;
        win[30] = (sn < 1024) ? bf2f(cin[((size_t)b * 1024 + sn) * 1024 + d]) : 0.f;
    }
}

// ============================ host orchestration =============================
extern "C" void kernel_launch(void* const* d_in, const int* in_sizes, int n_in,
                              void* d_out, int out_size, void* d_ws, size_t ws_size,
                              hipStream_t stream) {
    (void)in_sizes; (void)n_in; (void)out_size; (void)ws_size;
    const float* x         = (const float*)d_in[0];
    const float* ffn1_ln_g = (const float*)d_in[1];
    const float* ffn1_ln_b = (const float*)d_in[2];
    const float* ffn1_w1   = (const float*)d_in[3];
    const float* ffn1_b1   = (const float*)d_in[4];
    const float* ffn1_w2   = (const float*)d_in[5];
    const float* ffn1_b2   = (const float*)d_in[6];
    const float* attn_ln_g = (const float*)d_in[7];
    const float* attn_ln_b = (const float*)d_in[8];
    const float* qkv_w     = (const float*)d_in[9];
    const float* qkv_b     = (const float*)d_in[10];
    const float* out_w     = (const float*)d_in[11];
    const float* out_b     = (const float*)d_in[12];
    const float* dist_emb  = (const float*)d_in[13];
    const float* conv_ln_g = (const float*)d_in[14];
    const float* conv_ln_b = (const float*)d_in[15];
    const float* pw1_w     = (const float*)d_in[16];
    const float* dw_w      = (const float*)d_in[17];
    const float* dw_ln_g   = (const float*)d_in[18];
    const float* dw_ln_b   = (const float*)d_in[19];
    const float* pw2_w     = (const float*)d_in[20];
    const float* ffn2_ln_g = (const float*)d_in[21];
    const float* ffn2_ln_b = (const float*)d_in[22];
    const float* ffn2_w1   = (const float*)d_in[23];
    const float* ffn2_b1   = (const float*)d_in[24];
    const float* ffn2_w2   = (const float*)d_in[25];
    const float* ffn2_b2   = (const float*)d_in[26];
    const float* fin_ln_g  = (const float*)d_in[27];
    const float* fin_ln_b  = (const float*)d_in[28];

    char* ws = (char*)d_ws;
    constexpr size_t MB = 1024ull * 1024ull;
    // bf16 weights (N-major [N][K]) -- 46 MB
    u16* Wf1a = (u16*)(ws + 0 * MB);    // [4096][1024]
    u16* Wf2a = (u16*)(ws + 8 * MB);    // [1024][4096]
    u16* Wqkv = (u16*)(ws + 16 * MB);   // [3072][1024]
    u16* Wout = (u16*)(ws + 22 * MB);   // [1024][1024]
    u16* Wpw1 = (u16*)(ws + 24 * MB);   // [2048][1024] (already N-major)
    u16* Wpw2 = (u16*)(ws + 28 * MB);   // [1024][1024] (already N-major)
    u16* Wf1b = (u16*)(ws + 30 * MB);   // [4096][1024]
    u16* Wf2b = (u16*)(ws + 38 * MB);   // [1024][4096]
    u16* Wpe  = (u16*)(ws + 46 * MB);   // [129][64]
    // activations
    float* hA   = (float*)(ws + 48 * MB);   // residual stream A (32 MB)
    float* hB   = (float*)d_out;            // residual stream B aliases output
    char*  X1   = ws + 80 * MB;             // 64 MB: FFN intermediate / GLU pre-act
    char*  X2   = ws + 144 * MB;            // 48 MB: q/k/v, later c2 + dwconv out
    char*  X3   = ws + 192 * MB;            // 16 MB: bf16 LN outputs / attn out / y2
    u16*   t1   = (u16*)X1;
    float* glu  = (float*)X1;
    u16*   qbuf = (u16*)X2;
    u16*   kbuf = (u16*)(X2 + 16 * MB);
    u16*   vbuf = (u16*)(X2 + 32 * MB);
    u16*   c2   = (u16*)X2;                 // reuses q slot (dead by then)
    float* ybuf = (float*)(X2 + 16 * MB);
    u16*   lnb  = (u16*)X3;

    // ---- weight conversion (deterministic, every call) ----
    convert_transpose_kernel<<<(4096 * 1024) / 256, 256, 0, stream>>>(ffn1_w1, Wf1a, 1024, 4096);
    convert_transpose_kernel<<<(1024 * 4096) / 256, 256, 0, stream>>>(ffn1_w2, Wf2a, 4096, 1024);
    convert_transpose_kernel<<<(3072 * 1024) / 256, 256, 0, stream>>>(qkv_w,   Wqkv, 1024, 3072);
    convert_transpose_kernel<<<(1024 * 1024) / 256, 256, 0, stream>>>(out_w,   Wout, 1024, 1024);
    convert_kernel<<<(2048 * 1024) / 256, 256, 0, stream>>>(pw1_w, Wpw1, (size_t)2048 * 1024);
    convert_kernel<<<(1024 * 1024) / 256, 256, 0, stream>>>(pw2_w, Wpw2, (size_t)1024 * 1024);
    convert_kernel<<<(129 * 64 + 255) / 256, 256, 0, stream>>>(dist_emb, Wpe, (size_t)129 * 64);
    convert_transpose_kernel<<<(4096 * 1024) / 256, 256, 0, stream>>>(ffn2_w1, Wf1b, 1024, 4096);
    convert_transpose_kernel<<<(1024 * 4096) / 256, 256, 0, stream>>>(ffn2_w2, Wf2b, 4096, 1024);

    const int M = 8192;
    auto gemm = [&](const u16* A, const u16* Bw, const float* bias, const float* resid,
                    float* outf, u16* outh, u16* oq, u16* ok, u16* ov,
                    int N, int K, int act, int mode, float alpha) {
        GemmArgs ga;
        ga.A = A; ga.Bw = Bw; ga.bias = bias; ga.resid = resid;
        ga.outf = outf; ga.outh = outh; ga.oq = oq; ga.ok = ok; ga.ov = ov;
        ga.M = M; ga.N = N; ga.K = K; ga.act = act; ga.mode = mode; ga.alpha = alpha;
        dim3 grid((N + 127) / 128, M / 128);
        gemm_bf16_kernel<<<grid, 256, 0, stream>>>(ga);
    };

    // ---- FFN1 (half-step residual) ----
    ln_kernel<<<M, 256, 0, stream>>>(x, ffn1_ln_g, ffn1_ln_b, lnb, 0, 1);
    gemm(lnb, Wf1a, ffn1_b1, nullptr, nullptr, t1, nullptr, nullptr, nullptr, 4096, 1024, 1, 1, 1.0f);
    gemm(t1,  Wf2a, ffn1_b2, x, hA, nullptr, nullptr, nullptr, nullptr, 1024, 4096, 0, 0, 0.5f);

    // ---- self-attention with relative-position bias ----
    ln_kernel<<<M, 256, 0, stream>>>(hA, attn_ln_g, attn_ln_b, lnb, 0, 1);
    gemm(lnb, Wqkv, qkv_b, nullptr, nullptr, nullptr, qbuf, kbuf, vbuf, 3072, 1024, 0, 2, 1.0f);
    flash_attn_kernel<<<dim3(16, 128), 128, 0, stream>>>(qbuf, kbuf, vbuf, Wpe, lnb); // lnb = attn out
    gemm(lnb, Wout, out_b, hA, hB, nullptr, nullptr, nullptr, nullptr, 1024, 1024, 0, 0, 1.0f);

    // ---- conformer conv module ----
    ln_kernel<<<M, 256, 0, stream>>>(hB, conv_ln_g, conv_ln_b, lnb, 0, 1);
    gemm(lnb, Wpw1, nullptr, nullptr, glu, nullptr, nullptr, nullptr, nullptr, 2048, 1024, 0, 0, 1.0f);
    glu_kernel<<<(M * 1024) / 256, 256, 0, stream>>>(glu, c2);
    dwconv_kernel<<<dim3(4, 8, 8), 256, 0, stream>>>(c2, dw_w, ybuf);
    ln_kernel<<<M, 256, 0, stream>>>(ybuf, dw_ln_g, dw_ln_b, lnb, 1, 1);   // LN + SiLU
    gemm(lnb, Wpw2, nullptr, hB, hA, nullptr, nullptr, nullptr, nullptr, 1024, 1024, 0, 0, 1.0f);

    // ---- FFN2 (half-step residual) + final LN (in place on d_out) ----
    ln_kernel<<<M, 256, 0, stream>>>(hA, ffn2_ln_g, ffn2_ln_b, lnb, 0, 1);
    gemm(lnb, Wf1b, ffn2_b1, nullptr, nullptr, t1, nullptr, nullptr, nullptr, 4096, 1024, 1, 1, 1.0f);
    gemm(t1,  Wf2b, ffn2_b2, hA, hB, nullptr, nullptr, nullptr, nullptr, 1024, 4096, 0, 0, 0.5f);
    ln_kernel<<<M, 256, 0, stream>>>(hB, fin_ln_g, fin_ln_b, (void*)hB, 0, 0);
}
